// TransformerBlockDecoder_66056597012787
// MI455X (gfx1250) — compile-verified, matrix operand order checked by reading, not run on hardware
//
#include <hip/hip_runtime.h>
#include <hip/hip_bf16.h>
#include <stdint.h>

#define D_MODEL 1024
#define NHEADS  16
#define DHEAD   64
#define BATCH   2
#define SEQ     2048
#define EPS_LN  1e-5f

typedef __attribute__((ext_vector_type(16))) __bf16 v16bf;
typedef __attribute__((ext_vector_type(8)))  float  v8f;
typedef unsigned short bf16u;

__device__ __forceinline__ bf16u f2b(float f) {
  union { float f; uint32_t u; } c; c.f = f;
  uint32_t u = c.u;
  uint32_t r = u + 0x7FFFu + ((u >> 16) & 1u);   // round-to-nearest-even
  return (bf16u)(r >> 16);
}

// ---------------------------------------------------------------------------
// CDNA5 async global->LDS DMA (ASYNCcnt path). LDS destination is the low 32
// bits of the flat shared-memory address (LDS byte offset); global address is
// a per-lane 64-bit VGPR pair. 16 bytes per lane per issue.
// ---------------------------------------------------------------------------
__device__ __forceinline__ void async_copy_b128(uint32_t lds_byte_addr, uint64_t gaddr) {
  asm volatile("global_load_async_to_lds_b128 %0, %1, off"
               :
               : "v"(lds_byte_addr), "v"(gaddr)
               : "memory");
}
__device__ __forceinline__ void wait_asynccnt0() {
  asm volatile("s_wait_asynccnt 0x0" ::: "memory");
}

// ---------------------------------------------------------------------------
// fp32 -> bf16 cast (weights)
// ---------------------------------------------------------------------------
__global__ void cast_f32_bf16(const float* __restrict__ in, bf16u* __restrict__ out, int n) {
  int i = blockIdx.x * blockDim.x + threadIdx.x;
  int stride = gridDim.x * blockDim.x;
  for (; i < n; i += stride) out[i] = f2b(in[i]);
}

// ---------------------------------------------------------------------------
// LayerNorm over D=1024, one block (8 waves) per row, bf16 output
// ---------------------------------------------------------------------------
__global__ __launch_bounds__(256) void layernorm_bf16(
    const float* __restrict__ x, const float* __restrict__ scale,
    const float* __restrict__ shift, bf16u* __restrict__ out) {
  const int row = blockIdx.x;
  const int t = threadIdx.x;
  const float4* xr = (const float4*)(x + (size_t)row * D_MODEL);
  float4 v = xr[t];
  float s = v.x + v.y + v.z + v.w;
  float q = v.x * v.x + v.y * v.y + v.z * v.z + v.w * v.w;
  for (int m = 1; m < 32; m <<= 1) {
    s += __shfl_xor(s, m, 32);
    q += __shfl_xor(q, m, 32);
  }
  __shared__ float ss[8], sq[8];
  const int wave = t >> 5, lane = t & 31;
  if (lane == 0) { ss[wave] = s; sq[wave] = q; }
  __syncthreads();
  float ts = 0.f, tq = 0.f;
  for (int i = 0; i < 8; ++i) { ts += ss[i]; tq += sq[i]; }
  const float mean = ts * (1.0f / D_MODEL);
  const float var  = tq * (1.0f / D_MODEL) - mean * mean;
  const float rstd = rsqrtf(var + EPS_LN);
  const float4 g = ((const float4*)scale)[t];
  const float4 b = ((const float4*)shift)[t];
  bf16u* o = out + (size_t)row * D_MODEL + 4 * t;
  o[0] = f2b(g.x * (v.x - mean) * rstd + b.x);
  o[1] = f2b(g.y * (v.y - mean) * rstd + b.y);
  o[2] = f2b(g.z * (v.z - mean) * rstd + b.z);
  o[3] = f2b(g.w * (v.w - mean) * rstd + b.w);
}

// ---------------------------------------------------------------------------
// Generic bf16 WMMA GEMM:  C[M,N] = A[M,K] (row-major) * W[N,K]^T
// Tile: BM=128, BN=128, BK=32. 256 threads = 8 waves in 2(M) x 4(N) grid;
// each wave owns a 64x32 subtile = 4x2 wmma 16x16 accumulators.
// Mainloop: double-buffered LDS fed by async global->LDS DMA so the HBM/L2
// fetch of slab k+1 overlaps the 8-WMMA burst of slab k; one barrier/iter.
// Epilogue: +bias, optional GELU, optional +residual; f32 and/or bf16 out.
// ---------------------------------------------------------------------------
__global__ __launch_bounds__(256) void gemm_bf16_wmma(
    const bf16u* __restrict__ A, const bf16u* __restrict__ W,
    const float* __restrict__ bias, const float* __restrict__ residual,
    float* __restrict__ Cf, bf16u* __restrict__ Cb,
    int M, int Nn, int K, int act) {
  __shared__ alignas(32) bf16u As[2][128 * 32];
  __shared__ alignas(32) bf16u Bs[2][128 * 32];

  const int t     = threadIdx.x;
  const int lane  = t & 31;
  const int wave  = t >> 5;
  const int waveM = wave >> 2;        // 0..1
  const int waveN = wave & 3;         // 0..3
  const int blockN = blockIdx.x * 128;
  const int blockM = blockIdx.y * 128;
  const int lr = lane & 15;
  const int hi = lane >> 4;

  v8f acc[4][2];
  for (int i = 0; i < 4; ++i)
    for (int j = 0; j < 2; ++j)
      for (int r = 0; r < 8; ++r) acc[i][j][r] = 0.f;

  const int ldr = t >> 1;             // 0..127 (tile row)
  const int ldc = (t & 1) * 16;       // 0 or 16 (k-offset)
  const size_t aRow = (size_t)(blockM + ldr) * K;
  const size_t bRow = (size_t)(blockN + ldr) * K;

  const uint32_t aLds = (uint32_t)(uintptr_t)&As[0][ldr * 32 + ldc];
  const uint32_t bLds = (uint32_t)(uintptr_t)&Bs[0][ldr * 32 + ldc];
  const uint32_t bufStride = 128 * 32 * sizeof(bf16u);   // 8 KB

  auto issue = [&](int buf, int k0) {
    const uint64_t ga = (uint64_t)(uintptr_t)(A + aRow + k0 + ldc);
    const uint64_t gb = (uint64_t)(uintptr_t)(W + bRow + k0 + ldc);
    const uint32_t la = aLds + (uint32_t)buf * bufStride;
    const uint32_t lb = bLds + (uint32_t)buf * bufStride;
    async_copy_b128(la,      ga);
    async_copy_b128(la + 16, ga + 16);
    async_copy_b128(lb,      gb);
    async_copy_b128(lb + 16, gb + 16);
  };

  const int nk = K >> 5;
  issue(0, 0);
  wait_asynccnt0();
  __syncthreads();

  for (int ki = 0; ki < nk; ++ki) {
    const int buf = ki & 1;
    if (ki + 1 < nk) issue(buf ^ 1, (ki + 1) << 5);   // DMA next slab under WMMA

    const int khalf = hi * 16;
    v16bf afr[4], bfr[2];
    for (int mt = 0; mt < 4; ++mt) {
      int rr = waveM * 64 + mt * 16 + lr;
      afr[mt] = *(const v16bf*)&As[buf][rr * 32 + khalf];
    }
    for (int nt = 0; nt < 2; ++nt) {
      int cc = waveN * 32 + nt * 16 + lr;
      bfr[nt] = *(const v16bf*)&Bs[buf][cc * 32 + khalf];
    }
    for (int mt = 0; mt < 4; ++mt)
      for (int nt = 0; nt < 2; ++nt)
        acc[mt][nt] = __builtin_amdgcn_wmma_f32_16x16x32_bf16(
            false, afr[mt], false, bfr[nt], (short)0, acc[mt][nt], false, false);

    wait_asynccnt0();    // our DMA for the next slab landed
    __syncthreads();     // everyone done reading buf / writes visible
  }

  for (int mt = 0; mt < 4; ++mt) {
    for (int nt = 0; nt < 2; ++nt) {
      for (int r = 0; r < 8; ++r) {
        const int row = blockM + waveM * 64 + mt * 16 + r + 8 * hi;
        const int col = blockN + waveN * 32 + nt * 16 + lr;
        float v = acc[mt][nt][r];
        if (bias) v += bias[col];
        if (act) {  // tanh-GELU
          float u = v;
          v = 0.5f * u * (1.0f + tanhf(0.7978845608f * (u + 0.044715f * u * u * u)));
        }
        if (residual) v += residual[(size_t)row * Nn + col];
        if (Cf) Cf[(size_t)row * Nn + col] = v;
        if (Cb) Cb[(size_t)row * Nn + col] = f2b(v);
      }
    }
  }
}

// ---------------------------------------------------------------------------
// Flash attention (causal), bf16 in/out, f32 softmax state.
// Grid: (SEQ/128, BATCH*NHEADS). Block: 256 threads = 8 waves; each wave owns
// 16 query rows. K tiles arrive via async global->LDS DMA; V is transposed
// into LDS by hand (DMA cannot transpose). S = Q K^T and O += P V via WMMA.
// ---------------------------------------------------------------------------
__global__ __launch_bounds__(256) void flash_attn_wmma(
    const bf16u* __restrict__ Q, const bf16u* __restrict__ Kx,
    const bf16u* __restrict__ V, bf16u* __restrict__ O) {
  __shared__ alignas(32) bf16u Ks[64 * 64];
  __shared__ alignas(32) bf16u Vt[64 * 64];
  __shared__ alignas(32) bf16u Ps[8][16 * 64];

  const int t    = threadIdx.x;
  const int lane = t & 31;
  const int wave = t >> 5;
  const int bh   = blockIdx.y;
  const int b    = bh / NHEADS;
  const int h    = bh % NHEADS;
  const int q0   = blockIdx.x * 128;
  const int qrow = q0 + wave * 16;
  const int lr = lane & 15;
  const int hi = lane >> 4;

  // Q fragments, resident for the whole KV loop
  v16bf qf[2];
  {
    const size_t base = ((size_t)(b * SEQ + qrow + lr)) * D_MODEL + h * DHEAD;
    qf[0] = *(const v16bf*)(Q + base + hi * 16);
    qf[1] = *(const v16bf*)(Q + base + 32 + hi * 16);
  }

  float m[8], l[8];
  v8f o[4];
  for (int r = 0; r < 8; ++r) { m[r] = -INFINITY; l[r] = 0.f; }
  for (int dt = 0; dt < 4; ++dt)
    for (int r = 0; r < 8; ++r) o[dt][r] = 0.f;

  const int ldr = t >> 2;           // 0..63 key row
  const int ldc = (t & 3) * 16;     // 0..48 dh offset
  const uint32_t kLds = (uint32_t)(uintptr_t)&Ks[ldr * 64 + ldc];

  for (int kv0 = 0; kv0 <= q0 + 127; kv0 += 64) {
    {
      const size_t base = ((size_t)(b * SEQ + kv0 + ldr)) * D_MODEL + h * DHEAD + ldc;
      const uint64_t gk = (uint64_t)(uintptr_t)(Kx + base);
      async_copy_b128(kLds,      gk);         // K tile via ASYNCcnt DMA
      async_copy_b128(kLds + 16, gk + 16);
      const bf16u* vg = V + base;
      #pragma unroll
      for (int j = 0; j < 16; ++j) Vt[(ldc + j) * 64 + ldr] = vg[j];  // transpose
    }
    wait_asynccnt0();
    __syncthreads();

    // S tile: 16x64 per wave = 4 wmma tiles, K-dim = DHEAD = 2x32
    v8f s[4];
    for (int nt = 0; nt < 4; ++nt) {
      v16bf kf0 = *(const v16bf*)&Ks[(nt * 16 + lr) * 64 + hi * 16];
      v16bf kf1 = *(const v16bf*)&Ks[(nt * 16 + lr) * 64 + 32 + hi * 16];
      v8f z;
      for (int r = 0; r < 8; ++r) z[r] = 0.f;
      z = __builtin_amdgcn_wmma_f32_16x16x32_bf16(false, qf[0], false, kf0, (short)0, z, false, false);
      z = __builtin_amdgcn_wmma_f32_16x16x32_bf16(false, qf[1], false, kf1, (short)0, z, false, false);
      s[nt] = z;
    }

    // scale (1/sqrt(64)) + causal mask
    for (int nt = 0; nt < 4; ++nt)
      for (int r = 0; r < 8; ++r) {
        const int col = kv0 + nt * 16 + lr;
        const int row = qrow + r + 8 * hi;
        const float v = s[nt][r] * 0.125f;
        s[nt][r] = (col > row) ? -INFINITY : v;
      }

    // online softmax; shuffle masks 1..8 stay inside each 16-lane row group
    float alpha[8];
    for (int r = 0; r < 8; ++r) {
      float mx = fmaxf(fmaxf(s[0][r], s[1][r]), fmaxf(s[2][r], s[3][r]));
      for (int k = 1; k < 16; k <<= 1) mx = fmaxf(mx, __shfl_xor(mx, k, 32));
      const float nm = fmaxf(m[r], mx);
      alpha[r] = (m[r] == -INFINITY) ? 0.f : __expf(m[r] - nm);
      m[r] = nm;
      float rs = 0.f;
      for (int nt = 0; nt < 4; ++nt) {
        const float p = __expf(s[nt][r] - nm);
        s[nt][r] = p;
        rs += p;
      }
      for (int k = 1; k < 16; k <<= 1) rs += __shfl_xor(rs, k, 32);
      l[r] = l[r] * alpha[r] + rs;
    }
    for (int dt = 0; dt < 4; ++dt)
      for (int r = 0; r < 8; ++r) o[dt][r] *= alpha[r];

    // P -> per-wave LDS (C-layout -> A-layout re-fragmentation); same-wave DS
    // ops are in order, so no workgroup barrier is required here.
    for (int nt = 0; nt < 4; ++nt)
      for (int r = 0; r < 8; ++r)
        Ps[wave][(r + 8 * hi) * 64 + nt * 16 + lr] = f2b(s[nt][r]);

    v16bf pf0 = *(const v16bf*)&Ps[wave][lr * 64 + hi * 16];
    v16bf pf1 = *(const v16bf*)&Ps[wave][lr * 64 + 32 + hi * 16];

    for (int dt = 0; dt < 4; ++dt) {
      v16bf vf0 = *(const v16bf*)&Vt[(dt * 16 + lr) * 64 + hi * 16];
      v16bf vf1 = *(const v16bf*)&Vt[(dt * 16 + lr) * 64 + 32 + hi * 16];
      o[dt] = __builtin_amdgcn_wmma_f32_16x16x32_bf16(false, pf0, false, vf0, (short)0, o[dt], false, false);
      o[dt] = __builtin_amdgcn_wmma_f32_16x16x32_bf16(false, pf1, false, vf1, (short)0, o[dt], false, false);
    }
    __syncthreads();   // protect Ks/Vt before next tile load
  }

  for (int dt = 0; dt < 4; ++dt)
    for (int r = 0; r < 8; ++r) {
      const int row = qrow + r + 8 * hi;
      const float inv = (l[r] > 0.f) ? 1.0f / l[r] : 0.f;
      O[((size_t)(b * SEQ + row)) * D_MODEL + h * DHEAD + dt * 16 + lr] =
          f2b(o[dt][r] * inv);
    }
}

// ---------------------------------------------------------------------------
// Host-side orchestration
// ---------------------------------------------------------------------------
extern "C" void kernel_launch(void* const* d_in, const int* in_sizes, int n_in,
                              void* d_out, int out_size, void* d_ws, size_t ws_size,
                              hipStream_t stream) {
  (void)in_sizes; (void)n_in; (void)out_size; (void)ws_size;
  const float* x  = (const float*)d_in[0];
  const float* Wq = (const float*)d_in[1];
  const float* Wk = (const float*)d_in[2];
  const float* Wv = (const float*)d_in[3];
  const float* Wo = (const float*)d_in[4];
  const float* bo = (const float*)d_in[5];
  const float* g1 = (const float*)d_in[6];
  const float* s1 = (const float*)d_in[7];
  const float* g2 = (const float*)d_in[8];
  const float* s2 = (const float*)d_in[9];
  const float* W1 = (const float*)d_in[10];
  const float* b1 = (const float*)d_in[11];
  const float* W2 = (const float*)d_in[12];
  const float* b2 = (const float*)d_in[13];
  float* out = (float*)d_out;

  char* ws = (char*)d_ws;
  size_t off = 0;
  auto alloc = [&](size_t bytes) -> void* {
    off = (off + 255) & ~(size_t)255;
    void* p = ws + off;
    off += bytes;
    return p;
  };

  const size_t DD  = (size_t)D_MODEL * D_MODEL;      // 1 Mi
  const size_t MD  = (size_t)BATCH * SEQ * D_MODEL;  // 4 Mi
  const size_t MD4 = MD * 4;                         // 16 Mi

  bf16u* wqb = (bf16u*)alloc(DD * 2);
  bf16u* wkb = (bf16u*)alloc(DD * 2);
  bf16u* wvb = (bf16u*)alloc(DD * 2);
  bf16u* wob = (bf16u*)alloc(DD * 2);
  bf16u* w1b = (bf16u*)alloc(DD * 4 * 2);
  bf16u* w2b = (bf16u*)alloc(DD * 4 * 2);
  bf16u* hb  = (bf16u*)alloc(MD * 2);     // LN1(x)
  bf16u* qb  = (bf16u*)alloc(MD * 2);
  bf16u* kb  = (bf16u*)alloc(MD * 2);
  bf16u* vb  = (bf16u*)alloc(MD * 2);
  bf16u* pvb = (bf16u*)alloc(MD * 2);
  float* x1f = (float*)alloc(MD * 4);     // x + attn (residual 1)
  bf16u* h2b = (bf16u*)alloc(MD * 2);     // LN2(x1)
  bf16u* fb  = (bf16u*)alloc(MD4 * 2);    // gelu(h2 @ W1^T + b1)

  const int M = BATCH * SEQ;              // 4096

  // 1) weights -> bf16
  cast_f32_bf16<<<4096, 256, 0, stream>>>(Wq, wqb, (int)DD);
  cast_f32_bf16<<<4096, 256, 0, stream>>>(Wk, wkb, (int)DD);
  cast_f32_bf16<<<4096, 256, 0, stream>>>(Wv, wvb, (int)DD);
  cast_f32_bf16<<<4096, 256, 0, stream>>>(Wo, wob, (int)DD);
  cast_f32_bf16<<<8192, 256, 0, stream>>>(W1, w1b, (int)(DD * 4));
  cast_f32_bf16<<<8192, 256, 0, stream>>>(W2, w2b, (int)(DD * 4));

  // 2) LN1
  layernorm_bf16<<<M, 256, 0, stream>>>(x, g1, s1, hb);

  // 3) q/k/v projections
  dim3 gQKV(D_MODEL / 128, M / 128);
  gemm_bf16_wmma<<<gQKV, 256, 0, stream>>>(hb, wqb, nullptr, nullptr, nullptr, qb,
                                           M, D_MODEL, D_MODEL, 0);
  gemm_bf16_wmma<<<gQKV, 256, 0, stream>>>(hb, wkb, nullptr, nullptr, nullptr, kb,
                                           M, D_MODEL, D_MODEL, 0);
  gemm_bf16_wmma<<<gQKV, 256, 0, stream>>>(hb, wvb, nullptr, nullptr, nullptr, vb,
                                           M, D_MODEL, D_MODEL, 0);

  // 4) causal flash attention
  dim3 gAtt(SEQ / 128, BATCH * NHEADS);
  flash_attn_wmma<<<gAtt, 256, 0, stream>>>(qb, kb, vb, pvb);

  // 5) output projection + bias + residual -> x1 (f32)
  gemm_bf16_wmma<<<gQKV, 256, 0, stream>>>(pvb, wob, bo, x, x1f, nullptr,
                                           M, D_MODEL, D_MODEL, 0);

  // 6) LN2
  layernorm_bf16<<<M, 256, 0, stream>>>(x1f, g2, s2, h2b);

  // 7) FFN up + GELU (bf16 out for chaining)
  dim3 gF1(4 * D_MODEL / 128, M / 128);
  gemm_bf16_wmma<<<gF1, 256, 0, stream>>>(h2b, w1b, b1, nullptr, nullptr, fb,
                                          M, 4 * D_MODEL, D_MODEL, 1);

  // 8) FFN down + bias + residual -> final output (f32)
  gemm_bf16_wmma<<<gQKV, 256, 0, stream>>>(fb, w2b, b2, x1f, out, nullptr,
                                           M, D_MODEL, 4 * D_MODEL, 0);
}
